// VectorQuantizer_21603685499699
// MI455X (gfx1250) — matrix operand order repaired and checked by reading, not compiled, not run on hardware
//
#include <hip/hip_runtime.h>
#include <stdint.h>

// ---------------------------------------------------------------------------
// VQ-VAE vector quantizer for MI455X (gfx1250, wave32)
//   x:        [32, 64, 64, 64] f32  (B, D, H, W), N = 131072 vectors of D=64
//   codebook: [512, 64] f32
//   out = quantized [B,D,H,W] | indices(f32) [N] | dict_loss | commit_loss
// Core: V_WMMA_F32_16X16X32_BF16 computes X(16x64) . C^T(64x16) per code tile.
// Argmin trick: the 9-bit code index is embedded in the low mantissa bits of
// the f32 distance, so argmin reduces with plain v_min_f32 (branchless).
// ---------------------------------------------------------------------------

typedef __attribute__((ext_vector_type(16))) __bf16          v16bf;
typedef __attribute__((ext_vector_type(16))) unsigned short  v16us;
typedef __attribute__((ext_vector_type(8)))  float           v8f;

#define KCODES 512
#define DDIM   64
#define LDS_STRIDE 68   // u16 units (136 bytes): odd-dword-pair pad for ds_load_b64 bank spread

__device__ __forceinline__ unsigned short f2bf(float f) {
    union { __bf16 b; unsigned short s; } u;
    u.b = (__bf16)f;                       // native v_cvt (pairs into v_cvt_pk_bf16_f32)
    return u.s;
}

__device__ __forceinline__ float embed_idx(float dist, int code) {
    unsigned u = __float_as_uint(dist);
    u = (u & 0xFFFFFE00u) | (unsigned)code;   // v_bfi-able: low 9 bits <- code
    return __uint_as_float(u);
}

// ---- precompute |c_k|^2 into ws, zero the loss accumulator --------------------
__global__ __launch_bounds__(512)
void vq_pre(const float* __restrict__ cb, float* __restrict__ wsC2,
            float* __restrict__ wsLoss) {
    int k = blockIdx.x * blockDim.x + threadIdx.x;
    if (k == 0) wsLoss[0] = 0.0f;
    if (k < KCODES) {
        float s = 0.0f;
        #pragma unroll 8
        for (int d = 0; d < DDIM; ++d) {
            float c = cb[k * DDIM + d];
            s = fmaf(c, c, s);
        }
        wsC2[k] = s;
    }
}

// ---- main kernel --------------------------------------------------------------
__global__ __launch_bounds__(256)
void vq_main(const float* __restrict__ x, const float* __restrict__ cb,
             const float* __restrict__ wsC2,
             float* __restrict__ out, float* __restrict__ idxOut,
             float* __restrict__ wsLoss, int nPix) {
    __shared__ unsigned short sCB[KCODES * LDS_STRIDE];  // bf16 codebook, padded rows
    __shared__ float          sC2[KCODES];

    const int tid  = threadIdx.x;
    const int lane = tid & 31;

    // Stage bf16 codebook + |c|^2 into LDS (reused by all tiles of this block).
    for (int i = tid; i < KCODES * DDIM; i += 256) {
        int r = i >> 6, d = i & 63;
        sCB[r * LDS_STRIDE + d] = f2bf(cb[i]);
    }
    for (int i = tid; i < KCODES; i += 256) sC2[i] = wsC2[i];
    __syncthreads();

    const int half = lane >> 4;       // 0: lanes 0-15, 1: lanes 16-31
    const int m16  = lane & 15;       // row (A) / column (B,D) within tile
    const int kbA  = half * 16;       // A-matrix K base for this lane half
    const int offB = half * 16;       // B-matrix u16 offset within codebook row

    const int nTiles = nPix >> 4;
    const int waveId = blockIdx.x * 8 + (tid >> 5);
    const int nWaves = gridDim.x * 8;

    float lsum = 0.0f;

    for (int tile = waveId; tile < nTiles; tile += nWaves) {
        const int pixBase = tile << 4;
        const int bIdx    = pixBase >> 12;       // H*W = 4096 pixels per batch image
        const int pixInB  = pixBase & 4095;
        const float* xr = x + (size_t)bIdx * (DDIM * 4096) + pixInB + m16;

        // ---- A tile: 16 rows x 64 dims as bf16, contiguous k = 16*half + s (same as B) -----------
        union { v16us s; v16bf v; } A0, A1;
        #pragma unroll
        for (int j = 0; j < 8; ++j) {
            A0.s[j]     = f2bf(xr[(size_t)(kbA + j)      << 12]);
            A0.s[j + 8] = f2bf(xr[(size_t)(kbA + 8 + j) << 12]);
            A1.s[j]     = f2bf(xr[(size_t)(kbA + 32 + j) << 12]);
            A1.s[j + 8] = f2bf(xr[(size_t)(kbA + 40 + j) << 12]);
        }

        // best[r]: min over codes of (dist with low-9 mantissa bits = code index)
        float best[8];
        #pragma unroll
        for (int r = 0; r < 8; ++r) best[r] = 3.0e38f;

        // ---- sweep 32 code tiles of 16 codes each -----------------------------
        for (int t = 0; t < KCODES / 16; ++t) {
            const int code = t * 16 + m16;
            const unsigned short* bp = &sCB[code * LDS_STRIDE + offB];
            union { uint2 q[4]; v16bf v; } B0, B1;
            #pragma unroll
            for (int c = 0; c < 4; ++c) {
                B0.q[c] = *(const uint2*)(bp + c * 4);        // dims [offB .. offB+15]
                B1.q[c] = *(const uint2*)(bp + 32 + c * 4);   // dims [32+offB .. ]
            }

            v8f acc = {};
            acc = __builtin_amdgcn_wmma_f32_16x16x32_bf16(false, A0.v, false, B0.v,
                                                          (short)0, acc, false, false);
            acc = __builtin_amdgcn_wmma_f32_16x16x32_bf16(false, A1.v, false, B1.v,
                                                          (short)0, acc, false, false);

            const float c2v = sC2[code];
            #pragma unroll
            for (int r = 0; r < 8; ++r) {
                float dist = fmaf(-2.0f, acc[r], c2v);   // |c|^2 - 2 x.c
                best[r] = fminf(best[r], embed_idx(dist, code));
            }
        }

        // ---- branchless min-reduce across the 16 lanes of each half ----------
        #pragma unroll
        for (int r = 0; r < 8; ++r) {
            #pragma unroll
            for (int s = 1; s < 16; s <<= 1)
                best[r] = fminf(best[r], __shfl_xor(best[r], s, 32));
        }

        // ---- emit indices (rows 0-7 from lane 0, rows 8-15 from lane 16) -----
        if (m16 == 0) {
            #pragma unroll
            for (int r = 0; r < 8; ++r)
                idxOut[pixBase + half * 8 + r] =
                    (float)(__float_as_uint(best[r]) & 511u);
        }

        // ---- broadcast per-row winner so lane L serves pixel row (L&15) ------
        float myKey = 0.0f;
        #pragma unroll
        for (int r = 0; r < 8; ++r) {
            float v = __shfl(best[r], (m16 >> 3) << 4, 32);
            myKey = ((m16 & 7) == r) ? v : myKey;
        }
        const int myIdx = (int)(__float_as_uint(myKey) & 511u);

        // ---- gather code (f32), write quantized output, accumulate loss ------
        const float* qrow = cb + (size_t)myIdx * DDIM;
        float* ob = out + (size_t)bIdx * (DDIM * 4096) + pixInB + m16;
        #pragma unroll 4
        for (int it = 0; it < 32; ++it) {
            int d = it * 2 + half;           // lanes cover (d, pixel) coalesced
            float q = qrow[d];
            size_t off = (size_t)d << 12;
            float xv = xr[off];
            ob[off] = q;
            float diff = xv - q;
            lsum = fmaf(diff, diff, lsum);
        }
    }

    // ---- one atomic per wave for the loss sum --------------------------------
    #pragma unroll
    for (int s = 1; s < 32; s <<= 1) lsum += __shfl_xor(lsum, s, 32);
    if (lane == 0) atomicAdd(wsLoss, lsum);
}

// ---- finalize: both losses are numerically mean((x - q)^2) -------------------
__global__ void vq_fin(const float* __restrict__ wsLoss,
                       float* __restrict__ lossOut, float invN) {
    float m = wsLoss[0] * invN;
    lossOut[0] = m;   // dictionary_loss
    lossOut[1] = m;   // commitment_loss
}

extern "C" void kernel_launch(void* const* d_in, const int* in_sizes, int n_in,
                              void* d_out, int out_size, void* d_ws, size_t ws_size,
                              hipStream_t stream) {
    const float* x  = (const float*)d_in[0];
    const float* cb = (const float*)d_in[1];
    const int nElem = in_sizes[0];        // B*D*H*W = 8388608
    const int nPix  = nElem / DDIM;       // 131072

    float* wsC2   = (float*)d_ws;         // [512] |c|^2
    float* wsLoss = wsC2 + KCODES;        // [1] loss accumulator

    float* out     = (float*)d_out;       // [nElem] quantized, [B,D,H,W]
    float* idxOut  = out + nElem;         // [nPix] indices as f32
    float* lossOut = idxOut + nPix;       // [2] losses

    vq_pre <<<1, 512, 0, stream>>>(cb, wsC2, wsLoss);
    vq_main<<<256, 256, 0, stream>>>(x, cb, wsC2, out, idxOut, wsLoss, nPix);
    vq_fin <<<1, 1, 0, stream>>>(wsLoss, lossOut, 1.0f / (float)nElem);
}
